// RelationalLayer_10711648436521
// MI455X (gfx1250) — matrix operand order repaired and checked by reading, not compiled, not run on hardware
//
#include <hip/hip_runtime.h>

typedef _Float16 h16x16 __attribute__((ext_vector_type(16)));
typedef _Float16 h16x8  __attribute__((ext_vector_type(8)));
typedef float    f32x8  __attribute__((ext_vector_type(8)));

#define LDS_STRIDE 264  // 256 + 8 halfs padding (row = 528B = 33*16B, keeps 16B alignment)

// ---------------------------------------------------------------------------
// S := 0  (64 x 256 f32 partial-sum buffer; ws is poisoned, must re-zero each call)
// ---------------------------------------------------------------------------
__global__ void zero_s(float* __restrict__ S) {
  S[blockIdx.x * 256 + threadIdx.x] = 0.0f;
}

// ---------------------------------------------------------------------------
// A[b,j,n] = sum_k x[b,j,k] * W0[k,n]            (k = 0..25)
// B[b,j,n] = sum_k x[b,j,k] * W0[26+k,n] + b0[n]
// so that h0[b,i,j] = relu(A[b,j] + B[b,i])  -- layer 0 factorized:
// 7 GFLOP of pairwise layer-0 GEMM collapses to a 109 MFLOP precompute.
// grid: 64*64 blocks (one per (b,j) row), 256 threads (one per n).
// ---------------------------------------------------------------------------
__global__ void precompute_ab(const float* __restrict__ x, const float* __restrict__ W0,
                              const float* __restrict__ b0,
                              float* __restrict__ A, float* __restrict__ B) {
  __shared__ float xs[26];
  const int row = blockIdx.x;   // b*64 + j
  const int t   = threadIdx.x;  // n
  if (t < 26) xs[t] = x[row * 26 + t];
  __syncthreads();
  float a = 0.0f, bb = b0[t];
#pragma unroll
  for (int k = 0; k < 26; ++k) {
    a  += xs[k] * W0[k * 256 + t];
    bb += xs[k] * W0[(26 + k) * 256 + t];
  }
  A[row * 256 + t] = a;
  B[row * 256 + t] = bb;
}

// ---------------------------------------------------------------------------
// Wt[n][k] = (f16) W[k][n] for W1..W3 so that each lane's 16 B-fragment
// elements (K = base..base+15 at fixed N) are contiguous in memory.
// grid: 3*256 blocks, 256 threads.
// ---------------------------------------------------------------------------
__global__ void transpose_w(const float* __restrict__ W1, const float* __restrict__ W2,
                            const float* __restrict__ W3, _Float16* __restrict__ W1t,
                            _Float16* __restrict__ W2t, _Float16* __restrict__ W3t) {
  const int w = blockIdx.x >> 8;
  const int k = blockIdx.x & 255;
  const int n = threadIdx.x;
  const float*   src = (w == 0) ? W1 : (w == 1) ? W2 : W3;
  _Float16*      dst = (w == 0) ? W1t : (w == 1) ? W2t : W3t;
  dst[n * 256 + k] = (_Float16)src[k * 256 + n];
}

// ---------------------------------------------------------------------------
// Main fused pair-MLP. One block per (b, i): 64 pair-rows (j = 0..63),
// layers 1..3 as WMMA f16->f32 GEMMs, relu, then column-sum into S[b].
// 8 waves; wave w owns output columns [32w, 32w+32).
// ---------------------------------------------------------------------------
__global__ __launch_bounds__(256) void relnet_pairs(
    const float* __restrict__ A, const float* __restrict__ B,
    const _Float16* __restrict__ W1t, const _Float16* __restrict__ W2t,
    const _Float16* __restrict__ W3t,
    const float* __restrict__ b1, const float* __restrict__ b2,
    const float* __restrict__ b3, float* __restrict__ S) {
  __shared__ __align__(16) _Float16 Hs[64][LDS_STRIDE];

  const int t = threadIdx.x;
  const int b = blockIdx.x >> 6;
  const int i = blockIdx.x & 63;

  // ---- H0[j][n] = relu(A[b,j,n] + B[b,i,n]) -> f16 in LDS ----
  // Vectorized: float4 global loads, h16x8 (b128) LDS stores.
  {
    const int j  = t >> 2;            // row 0..63
    const int n0 = (t & 3) * 64;      // 64 contiguous cols per thread
    const float4* Ar = (const float4*)(A + ((size_t)(b * 64 + j)) * 256 + n0);
    const float4* Br = (const float4*)(B + ((size_t)(b * 64 + i)) * 256 + n0);
#pragma unroll
    for (int e = 0; e < 8; ++e) {     // 8 x (8 halfs) = 64 cols
      float4 a0 = Ar[2 * e],     b0v = Br[2 * e];
      float4 a1 = Ar[2 * e + 1], b1v = Br[2 * e + 1];
      h16x8 pk;
      pk[0] = (_Float16)fmaxf(a0.x + b0v.x, 0.0f);
      pk[1] = (_Float16)fmaxf(a0.y + b0v.y, 0.0f);
      pk[2] = (_Float16)fmaxf(a0.z + b0v.z, 0.0f);
      pk[3] = (_Float16)fmaxf(a0.w + b0v.w, 0.0f);
      pk[4] = (_Float16)fmaxf(a1.x + b1v.x, 0.0f);
      pk[5] = (_Float16)fmaxf(a1.y + b1v.y, 0.0f);
      pk[6] = (_Float16)fmaxf(a1.z + b1v.z, 0.0f);
      pk[7] = (_Float16)fmaxf(a1.w + b1v.w, 0.0f);
      *(h16x8*)&Hs[j][n0 + 8 * e] = pk;
    }
  }
  __syncthreads();

  const _Float16* Wts[3] = {W1t, W2t, W3t};
  const float*    bss[3] = {b1, b2, b3};

  const int wave = t >> 5;
  const int lane = t & 31;
  const int lhi  = (lane >> 4) & 1;  // 0: lanes 0-15, 1: lanes 16-31
  const int lmod = lane & 15;

  const int N0 = wave * 32 + lmod;  // column for n-tile 0
  const int N1 = N0 + 16;           // column for n-tile 1

  for (int layer = 0; layer < 3; ++layer) {
    f32x8 acc[4][2];
#pragma unroll
    for (int mt = 0; mt < 4; ++mt) { acc[mt][0] = {}; acc[mt][1] = {}; }

    const _Float16* Wt = Wts[layer];

#pragma unroll
    for (int kt = 0; kt < 8; ++kt) {
      // B fragment (32x16, f16): lane holds K = bk..bk+15 at column N, contiguous in Wt.
      const int bk = kt * 32 + lhi * 16;
      h16x16 bf0 = *(const h16x16*)(Wt + (size_t)N0 * 256 + bk);
      h16x16 bf1 = *(const h16x16*)(Wt + (size_t)N1 * 256 + bk);
      // A fragment (16x32, f16): lane holds row (lane&15), K = ak..ak+15 contiguous.
      const int ak = kt * 32 + lhi * 16;
#pragma unroll
      for (int mt = 0; mt < 4; ++mt) {
        const int row = mt * 16 + lmod;
        h16x8 lo = *(const h16x8*)&Hs[row][ak];
        h16x8 hi = *(const h16x8*)&Hs[row][ak + 8];
        h16x16 af = __builtin_shufflevector(lo, hi, 0, 1, 2, 3, 4, 5, 6, 7,
                                            8, 9, 10, 11, 12, 13, 14, 15);
        acc[mt][0] = __builtin_amdgcn_wmma_f32_16x16x32_f16(
            false, af, false, bf0, (short)0, acc[mt][0], false, false);
        acc[mt][1] = __builtin_amdgcn_wmma_f32_16x16x32_f16(
            false, af, false, bf1, (short)0, acc[mt][1], false, false);
      }
    }

    const float bias0 = bss[layer][N0];
    const float bias1 = bss[layer][N1];

    if (layer < 2) {
      // store relu(acc+bias) back into the (single) LDS H buffer in-place.
      __syncthreads();  // all reads of Hs complete before overwrite
#pragma unroll
      for (int mt = 0; mt < 4; ++mt) {
#pragma unroll
        for (int v = 0; v < 8; ++v) {
          const int row = mt * 16 + v + lhi * 8;  // C/D layout: M = v + 8*lhi
          Hs[row][N0] = (_Float16)fmaxf(acc[mt][0][v] + bias0, 0.0f);
          Hs[row][N1] = (_Float16)fmaxf(acc[mt][1][v] + bias1, 0.0f);
        }
      }
      __syncthreads();  // stores visible before next layer reads
    } else {
      // layer 3: relu then sum over the 64 pair-rows of this block.
      float s0 = 0.0f, s1 = 0.0f;
#pragma unroll
      for (int mt = 0; mt < 4; ++mt) {
#pragma unroll
        for (int v = 0; v < 8; ++v) {
          s0 += fmaxf(acc[mt][0][v] + bias0, 0.0f);
          s1 += fmaxf(acc[mt][1][v] + bias1, 0.0f);
        }
      }
      // fold lane halves: lanes l and l^16 hold M-sets {v} and {v+8} for same N
      s0 += __shfl_xor(s0, 16, 32);
      s1 += __shfl_xor(s1, 16, 32);
      if (lane < 16) {
        atomicAdd(&S[b * 256 + N0], s0);
        atomicAdd(&S[b * 256 + N1], s1);
      }
    }
  }
}

// ---------------------------------------------------------------------------
// Head: S (64x256) -> relu(L4) -> relu(L5) -> L6 -> out (64x32). All f32.
// grid: 64 blocks (one per batch row), 256 threads.
// ---------------------------------------------------------------------------
__global__ __launch_bounds__(256) void relnet_head(
    const float* __restrict__ S, const float* __restrict__ W4,
    const float* __restrict__ b4, const float* __restrict__ W5,
    const float* __restrict__ b5, const float* __restrict__ W6,
    const float* __restrict__ b6, float* __restrict__ out) {
  __shared__ float sh[2][256];
  const int b = blockIdx.x, t = threadIdx.x;
  sh[0][t] = S[b * 256 + t];
  __syncthreads();
  float acc = b4[t];
  for (int k = 0; k < 256; ++k) acc += sh[0][k] * W4[k * 256 + t];
  sh[1][t] = fmaxf(acc, 0.0f);
  __syncthreads();
  acc = b5[t];
  for (int k = 0; k < 256; ++k) acc += sh[1][k] * W5[k * 256 + t];
  __syncthreads();
  sh[0][t] = fmaxf(acc, 0.0f);
  __syncthreads();
  if (t < 32) {
    acc = b6[t];
    for (int k = 0; k < 256; ++k) acc += sh[0][k] * W6[k * 32 + t];
    out[b * 32 + t] = acc;
  }
}

// ---------------------------------------------------------------------------
extern "C" void kernel_launch(void* const* d_in, const int* in_sizes, int n_in,
                              void* d_out, int out_size, void* d_ws, size_t ws_size,
                              hipStream_t stream) {
  (void)in_sizes; (void)n_in; (void)out_size; (void)ws_size;
  const float* x  = (const float*)d_in[0];
  const float* W0 = (const float*)d_in[1];
  const float* b0 = (const float*)d_in[2];
  const float* W1 = (const float*)d_in[3];
  const float* b1 = (const float*)d_in[4];
  const float* W2 = (const float*)d_in[5];
  const float* b2 = (const float*)d_in[6];
  const float* W3 = (const float*)d_in[7];
  const float* b3 = (const float*)d_in[8];
  const float* W4 = (const float*)d_in[9];
  const float* b4 = (const float*)d_in[10];
  const float* W5 = (const float*)d_in[11];
  const float* b5 = (const float*)d_in[12];
  const float* W6 = (const float*)d_in[13];
  const float* b6 = (const float*)d_in[14];
  float* out = (float*)d_out;

  // workspace layout (total ~8.44 MB)
  char* ws = (char*)d_ws;
  float*    A   = (float*)(ws);                         // 64*64*256 f32 = 4 MB
  float*    Bv  = (float*)(ws + 4u * 1024 * 1024);      // 4 MB
  _Float16* W1t = (_Float16*)(ws + 8u * 1024 * 1024);   // 128 KB
  _Float16* W2t = W1t + 256 * 256;                      // 128 KB
  _Float16* W3t = W2t + 256 * 256;                      // 128 KB
  float*    S   = (float*)(ws + 8u * 1024 * 1024 + 3u * 256 * 256 * sizeof(_Float16));

  zero_s<<<64, 256, 0, stream>>>(S);
  precompute_ab<<<64 * 64, 256, 0, stream>>>(x, W0, b0, A, Bv);
  transpose_w<<<3 * 256, 256, 0, stream>>>(W1, W2, W3, W1t, W2t, W3t);
  relnet_pairs<<<64 * 64, 256, 0, stream>>>(A, Bv, W1t, W2t, W3t, b1, b2, b3, S);
  relnet_head<<<64, 256, 0, stream>>>(S, W4, b4, W5, b5, W6, b6, out);
}
